// RDAB_81879256531184
// MI455X (gfx1250) — matrix operand order repaired and checked by reading, not compiled, not run on hardware
//
#include <hip/hip_runtime.h>
#include <hip/hip_bf16.h>

typedef __attribute__((ext_vector_type(16))) _Float16 v16h;
typedef __attribute__((ext_vector_type(8)))  _Float16 v8h;
typedef __attribute__((ext_vector_type(8)))  float    v8f;

#define BB 16
#define CC 64
#define HH 128
#define WW 128
#define NPIX (BB*HH*WW)          // 262144
#define NELEM (BB*CC*HH*WW)      // 16777216

__device__ __forceinline__ float leakyf(float v) { return v > 0.f ? v : 0.1f * v; }

// ---------------------------------------------------------------------------
// Dynamic per-(b,c) 3x3 kernels:  hid = leaky(d @ k1);  kern = hid @ k2
// grid = 16 (one per batch), block = 64
// ---------------------------------------------------------------------------
__global__ void k_dynkern(const float* __restrict__ d,
                          const float* __restrict__ k1,   // [64,64] (in,out)
                          const float* __restrict__ k2,   // [64,576]
                          float* __restrict__ kern)       // [B*64][9]
{
    __shared__ float hid[CC];
    const int b = blockIdx.x;
    const int c = threadIdx.x;
    float s = 0.f;
    for (int j = 0; j < CC; ++j) s += d[b * CC + j] * k1[j * CC + c];
    hid[c] = leakyf(s);
    __syncthreads();
    #pragma unroll
    for (int t = 0; t < 9; ++t) {
        float a = 0.f;
        for (int j = 0; j < CC; ++j) a += hid[j] * k2[j * (CC * 9) + c * 9 + t];
        kern[(b * CC + c) * 9 + t] = a;
    }
}

// ---------------------------------------------------------------------------
// NCHW f32 -> NHWC f16 convert
// ---------------------------------------------------------------------------
__global__ void k_convert(const float* __restrict__ x, _Float16* __restrict__ xh)
{
    int i = blockIdx.x * blockDim.x + threadIdx.x;   // NHWC linear index
    int c = i & 63, w = (i >> 6) & 127, h = (i >> 13) & 127, b = (i >> 20) & 15;
    xh[i] = (_Float16)x[((b * CC + c) * HH + h) * WW + w];
}

// ---------------------------------------------------------------------------
// Pre-swizzle OIHW f32 conv weights into WMMA B-fragment layout (f16).
// frag = ((tap*2+kc)*4+nt); element [lane][j]:
//   oc = nt*16 + (lane&15), ic = kc*32 + (lane>>4)*16 + j
// ---------------------------------------------------------------------------
__global__ void k_prepw(const float* __restrict__ w, _Float16* __restrict__ bw)
{
    int gid = blockIdx.x * blockDim.x + threadIdx.x;
    if (gid >= 72 * 32) return;
    int lane = gid & 31, frag = gid >> 5;
    int nt = frag & 3, kc = (frag >> 2) & 1, tap = frag >> 3;
    int ky = tap / 3, kx = tap % 3;
    int oc = nt * 16 + (lane & 15);
    #pragma unroll
    for (int j = 0; j < 16; ++j) {
        int ic = kc * 32 + (lane >> 4) * 16 + j;
        bw[gid * 16 + j] = (_Float16)w[((oc * CC + ic) * 3 + ky) * 3 + kx];
    }
}

// ---------------------------------------------------------------------------
// Gate map: M[b,h,w] = sigmoid(sum_c cw[c]*xh[b,h,w,c] + cb)
// ---------------------------------------------------------------------------
__global__ void k_gate(const _Float16* __restrict__ xh,
                       const float* __restrict__ cw, const float* __restrict__ cb,
                       float* __restrict__ gm)
{
    int p = blockIdx.x * blockDim.x + threadIdx.x;   // pixel index [0, NPIX)
    const v8h* v = (const v8h*)&xh[(size_t)p * CC];
    float s = cb[0];
    #pragma unroll
    for (int j = 0; j < 8; ++j) {
        v8h q = v[j];
        #pragma unroll
        for (int e = 0; e < 8; ++e) s += cw[j * 8 + e] * (float)q[e];
    }
    gm[p] = 1.f / (1.f + __expf(-s));
}

// ---------------------------------------------------------------------------
// RDA elementwise: out = leaky( leaky(depthwise3x3(xh)) * M + xh )
// (outer leaky from the surrounding leaky(rda_conv(...)) is fused)
// ---------------------------------------------------------------------------
__global__ void k_rda(const _Float16* __restrict__ xh,
                      const float* __restrict__ gm,
                      const float* __restrict__ kern,
                      _Float16* __restrict__ out)
{
    int i = blockIdx.x * blockDim.x + threadIdx.x;
    int c = i & 63, w = (i >> 6) & 127, h = (i >> 13) & 127, b = (i >> 20) & 15;
    const float* kr = &kern[((b * CC) + c) * 9];
    float dw = 0.f;
    #pragma unroll
    for (int dy = -1; dy <= 1; ++dy) {
        #pragma unroll
        for (int dx = -1; dx <= 1; ++dx) {
            int h2 = h + dy, w2 = w + dx;
            float v = 0.f;
            if (h2 >= 0 && h2 < HH && w2 >= 0 && w2 < WW)
                v = (float)xh[(((size_t)(b * HH + h2) * WW + w2) * CC) + c];
            dw += kr[(dy + 1) * 3 + (dx + 1)] * v;
        }
    }
    float m  = gm[(b * HH + h) * WW + w];
    float xv = (float)xh[i];
    out[i] = (_Float16)leakyf(leakyf(dw) * m + xv);
}

// ---------------------------------------------------------------------------
// Implicit-GEMM 3x3 conv 64->64 with v_wmma_f32_16x16x32_f16.
// One block per (b, output row h). 8 waves; wave = 16 pixels x 64 oc.
// mode 0: y = leaky(conv + bias) -> f16 NHWC
// mode 1: y = conv + bias + x    -> f32 NCHW (d_out)
// ---------------------------------------------------------------------------
__global__ void __launch_bounds__(256)
k_conv(const _Float16* __restrict__ in,      // NHWC f16
       const _Float16* __restrict__ bw,      // swizzled B fragments [72][32][16]
       const float* __restrict__ bias,       // [64]
       const float* __restrict__ xres,       // NCHW f32 (mode 1) or nullptr
       _Float16* __restrict__ outh,          // NHWC f16 (mode 0)
       float* __restrict__ outf,             // NCHW f32 (mode 1)
       int mode)
{
    __shared__ _Float16 lds[3 * 130 * CC];   // 3 padded rows, ~49 KB

    const int b = blockIdx.x / HH;
    const int h = blockIdx.x % HH;

    // ---- stage padded 3-row window into LDS (8-half chunks, b128 ds ops) ----
    const int chunksPerRow = 130 * CC / 8;   // 1040
    for (int idx = threadIdx.x; idx < 3 * chunksPerRow; idx += blockDim.x) {
        int r   = idx / chunksPerRow;
        int rem = idx - r * chunksPerRow;
        int col = rem >> 3;                  // 0..129 -> w_in = col-1
        int sub = (rem & 7) * 8;
        int h_in = h + r - 1, w_in = col - 1;
        v8h val = {};
        if (h_in >= 0 && h_in < HH && w_in >= 0 && w_in < WW)
            val = *(const v8h*)&in[(((size_t)(b * HH + h_in) * WW + w_in) * CC) + sub];
        *(v8h*)&lds[r * (130 * CC) + col * CC + sub] = val;
    }

    // CDNA5 prefetch of the next row window (global_prefetch_b8)
    if (threadIdx.x == 0 && h + 2 < HH)
        __builtin_prefetch(&in[(size_t)(b * HH + h + 2) * WW * CC], 0, 1);

    __syncthreads();

    const int wave = threadIdx.x >> 5;
    const int lane = threadIdx.x & 31;
    const int w0   = wave * 16;
    const int m    = lane & 15;
    const int hi   = lane >> 4;

    v8f acc[4];
    #pragma unroll
    for (int nt = 0; nt < 4; ++nt) acc[nt] = (v8f){0.f,0.f,0.f,0.f,0.f,0.f,0.f,0.f};

    const v16h* bfr = (const v16h*)bw;

    #pragma unroll
    for (int tap = 0; tap < 9; ++tap) {
        const int r  = tap / 3;
        const int dc = tap % 3;              // lds col = w0+m+dc  (w_in = w+dx)
        #pragma unroll
        for (int kc = 0; kc < 2; ++kc) {
            // A fragment: lanes 0-15 hold K{0..15}, lanes 16-31 K{16..31}
            int base = r * (130 * CC) + (w0 + m + dc) * CC + kc * 32 + hi * 16;
            union { v16h v; v8h p[2]; } au;
            au.p[0] = *(const v8h*)&lds[base];
            au.p[1] = *(const v8h*)&lds[base + 8];
            #pragma unroll
            for (int nt = 0; nt < 4; ++nt) {
                v16h bv = bfr[(((tap * 2 + kc) * 4 + nt) * 32) + lane];
                acc[nt] = __builtin_amdgcn_wmma_f32_16x16x32_f16(
                    false, au.v, false, bv, (short)0, acc[nt], false, false);
            }
        }
    }

    // ---- epilogue ----
    #pragma unroll
    for (int nt = 0; nt < 4; ++nt) {
        const int oc = nt * 16 + (lane & 15);
        const float bi = bias[oc];
        #pragma unroll
        for (int r = 0; r < 8; ++r) {
            const int mm = r + 8 * hi;       // C/D layout: vgpr r -> M=r / M=8+r
            const int wp = w0 + mm;
            float v = acc[nt][r] + bi;
            if (mode == 0) {
                outh[(((size_t)(b * HH + h) * WW + wp) * CC) + oc] = (_Float16)leakyf(v);
            } else {
                size_t o = ((size_t)(b * CC + oc) * HH + h) * WW + wp;
                outf[o] = v + xres[o];
            }
        }
    }
}

// ---------------------------------------------------------------------------
extern "C" void kernel_launch(void* const* d_in, const int* in_sizes, int n_in,
                              void* d_out, int out_size, void* d_ws, size_t ws_size,
                              hipStream_t stream) {
    const float* x       = (const float*)d_in[0];
    const float* dvec    = (const float*)d_in[1];
    const float* da1_k1  = (const float*)d_in[2];
    const float* da1_k2  = (const float*)d_in[3];
    const float* da1_cw  = (const float*)d_in[4];
    const float* da1_cb  = (const float*)d_in[5];
    const float* da2_k1  = (const float*)d_in[6];
    const float* da2_k2  = (const float*)d_in[7];
    const float* da2_cw  = (const float*)d_in[8];
    const float* da2_cb  = (const float*)d_in[9];
    const float* conv1_w = (const float*)d_in[10];
    const float* conv1_b = (const float*)d_in[11];
    const float* conv2_w = (const float*)d_in[12];
    const float* conv2_b = (const float*)d_in[13];

    char* ws = (char*)d_ws;
    size_t oA  = 0;
    size_t oB  = oA  + (size_t)NELEM * 2;
    size_t oC  = oB  + (size_t)NELEM * 2;
    size_t oM  = oC  + (size_t)NELEM * 2;
    size_t oK1 = oM  + (size_t)NPIX * 4;
    size_t oK2 = oK1 + (size_t)BB * CC * 9 * 4;
    size_t oW1 = oK2 + (size_t)BB * CC * 9 * 4;
    size_t oW2 = oW1 + (size_t)72 * 32 * 16 * 2;

    _Float16* bufA = (_Float16*)(ws + oA);
    _Float16* bufB = (_Float16*)(ws + oB);
    _Float16* bufC = (_Float16*)(ws + oC);
    float*    gmap = (float*)(ws + oM);
    float*    kr1  = (float*)(ws + oK1);
    float*    kr2  = (float*)(ws + oK2);
    _Float16* bw1  = (_Float16*)(ws + oW1);
    _Float16* bw2  = (_Float16*)(ws + oW2);

    // prep
    k_prepw  <<<9, 256, 0, stream>>>(conv1_w, bw1);
    k_prepw  <<<9, 256, 0, stream>>>(conv2_w, bw2);
    k_dynkern<<<BB, CC, 0, stream>>>(dvec, da1_k1, da1_k2, kr1);
    k_dynkern<<<BB, CC, 0, stream>>>(dvec, da2_k1, da2_k2, kr2);
    k_convert<<<NELEM / 256, 256, 0, stream>>>(x, bufA);

    // stage A: t1 = leaky(rda_conv(x))
    k_gate<<<NPIX / 256, 256, 0, stream>>>(bufA, da1_cw, da1_cb, gmap);
    k_rda <<<NELEM / 256, 256, 0, stream>>>(bufA, gmap, kr1, bufB);

    // stage B: t2 = leaky(conv1(t1) + b1)   [WMMA]
    k_conv<<<BB * HH, 256, 0, stream>>>(bufB, bw1, conv1_b, nullptr, bufC, nullptr, 0);

    // stage C: t3 = leaky(rda_conv(t2))
    k_gate<<<NPIX / 256, 256, 0, stream>>>(bufC, da2_cw, da2_cb, gmap);
    k_rda <<<NELEM / 256, 256, 0, stream>>>(bufC, gmap, kr2, bufA);

    // stage D: out = conv2(t3) + b2 + x     [WMMA, f32 NCHW]
    k_conv<<<BB * HH, 256, 0, stream>>>(bufA, bw2, conv2_b, x, nullptr, (float*)d_out, 1);
}
